// ResNet_15461882266336
// MI455X (gfx1250) — matrix operand order repaired and checked by reading, not compiled, not run on hardware
//
#include <hip/hip_runtime.h>
#include <hip/hip_bf16.h>
#include <math.h>

// ---------------- types ----------------
typedef __attribute__((ext_vector_type(16))) _Float16 v16h;
typedef __attribute__((ext_vector_type(8)))  _Float16 v8h;
typedef __attribute__((ext_vector_type(4)))  _Float16 v4h;
typedef __attribute__((ext_vector_type(8)))  float    v8f;
typedef __attribute__((ext_vector_type(4)))  int      v4i;
typedef __attribute__((ext_vector_type(8)))  int      v8i;
typedef __attribute__((ext_vector_type(4)))  unsigned int v4u;

#define AS1 __attribute__((address_space(1)))
#define AS3 __attribute__((address_space(3)))

#if defined(__has_builtin)
#if __has_builtin(__builtin_amdgcn_global_load_async_to_lds_b128)
#define HAVE_ASYNC_LDS 1
#endif
#if __has_builtin(__builtin_amdgcn_tensor_load_to_lds)
#define HAVE_TDM 1
#endif
#endif

#define C_IN   256
#define K_OUT  256
#define HSZ    56
#define HW     3136          // 56*56
#define NBATCH 32
#define NTOT   (NBATCH*HW)   // 100352 spatial positions
#define XTOT   (NBATCH*C_IN*HW)  // 25690112 x elements
#define KSZ    2304          // 256*3*3
#define NGRP   147456        // KSZ*K_OUT/4
#define HALF_LVLS 127.0f
#define STEP      (1.0f/127.0f)
#define CMAX      63.5f      // M2D*HALF_LVLS
#define WLIM      63.5f      // (1-M2D)*HALF_LVLS

#define BM 128
#define BN 64
#define BK 32
#define NSTEPS (KSZ/BK)      // 72

// ---------------- init: stats + x zero-pad ----------------
__global__ void init_ws(float* stats, _Float16* xpad) {
  stats[0] = 0.f; stats[1] = 0.f;
  for (int i = 0; i < 16; ++i) xpad[i] = (_Float16)0.f;
}

// ---------------- weight pipeline ----------------
__device__ __forceinline__ void group_lv_cent(const float4 wv,
                                              float lv[4], float& cent) {
  lv[0] = fminf(fmaxf(rintf(wv.x * HALF_LVLS), -HALF_LVLS), HALF_LVLS);
  lv[1] = fminf(fmaxf(rintf(wv.y * HALF_LVLS), -HALF_LVLS), HALF_LVLS);
  lv[2] = fminf(fmaxf(rintf(wv.z * HALF_LVLS), -HALF_LVLS), HALF_LVLS);
  lv[3] = fminf(fmaxf(rintf(wv.w * HALF_LVLS), -HALF_LVLS), HALF_LVLS);
  float gm = (lv[0] + lv[1] + lv[2] + lv[3]) * 0.25f;
  cent = fminf(fmaxf(rintf(gm), -CMAX), CMAX);
}

__global__ __launch_bounds__(256) void w_stats(const float* __restrict__ w,
                                               float* __restrict__ stats) {
  int g = blockIdx.x * 256 + threadIdx.x;           // one group of 4 per thread
  float4 wv = ((const float4*)w)[g];
  float lv[4], cent;
  group_lv_cent(wv, lv, cent);
  float s1 = 0.f, s2 = 0.f;
#pragma unroll
  for (int i = 0; i < 4; ++i) {
    float d = lv[i] - cent;
    s1 += d; s2 += d * d;
  }
  __shared__ float r1[256], r2[256];
  r1[threadIdx.x] = s1; r2[threadIdx.x] = s2;
  __syncthreads();
  for (int s = 128; s > 0; s >>= 1) {
    if (threadIdx.x < s) {
      r1[threadIdx.x] += r1[threadIdx.x + s];
      r2[threadIdx.x] += r2[threadIdx.x + s];
    }
    __syncthreads();
  }
  if (threadIdx.x == 0) {
    atomicAdd(&stats[0], r1[0]);
    atomicAdd(&stats[1], r2[0]);
  }
}

__global__ __launch_bounds__(256) void w_final(const float* __restrict__ w,
                                               const float* __restrict__ cb,
                                               const int*   __restrict__ sig,
                                               const float* __restrict__ stats,
                                               _Float16*    __restrict__ wf) {
  int g = blockIdx.x * 256 + threadIdx.x;
  float4 wv = ((const float4*)w)[g];
  float lv[4], cent;
  group_lv_cent(wv, lv, cent);
  float d[4];
#pragma unroll
  for (int i = 0; i < 4; ++i) d[i] = lv[i] - cent;

  const float ntot = (float)(NGRP * 4);
  float mean = stats[0] / ntot;
  float var  = stats[1] / ntot - mean * mean;
  float scale = 4.0f * sqrtf(fmaxf(var, 0.f));      // CHECK_FACTOR * std

  float md = (d[0] + d[1] + d[2] + d[3]) * 0.25f;
  int best = 0;
  float bd = fabsf(md - scale * cb[0]);
#pragma unroll
  for (int j = 1; j < 4; ++j) {
    float dj = fabsf(md - scale * cb[j]);
    if (dj < bd) { bd = dj; best = j; }              // argmin keeps first min
  }
  float keep = (best == sig[g]) ? 1.f : 0.f;

  v4h hv;
#pragma unroll
  for (int i = 0; i < 4; ++i) {
    float dd  = d[i] * keep;                         // masked rescue
    float dev = rintf(fminf(fmaxf(dd, -WLIM), WLIM));
    hv[i] = (_Float16)((dev + cent) * STEP);
  }
  ((v4h*)wf)[g] = hv;
}

// ---------------- x -> f16 ----------------
__global__ __launch_bounds__(256) void x_to_f16(const float* __restrict__ x,
                                                _Float16* __restrict__ xf,
                                                int n4) {
  int i = blockIdx.x * 256 + threadIdx.x;
  if (i < n4) {
    float4 v = ((const float4*)x)[i];
    v4h h = { (_Float16)v.x, (_Float16)v.y, (_Float16)v.z, (_Float16)v.w };
    ((v4h*)xf)[i] = h;
  }
}

// ---------------- implicit-GEMM conv with WMMA ----------------
// M = 256 out channels, N = 100352 spatial, K = 2304 (c*9 + kh*3 + kw).
// Block: 256 threads = 8 waves (4x2), tile 128(M) x 64(N); wave = 32x32 (4 acc);
// K-step 32, double-buffered LDS. A tile staged by TDM (or async-to-LDS),
// B tile is a scalarized im2col gather with zero-pad select.
__global__ __launch_bounds__(256) void conv_wmma(const _Float16* __restrict__ xf,
                                                 const _Float16* __restrict__ wf,
                                                 float* __restrict__ out) {
  __shared__ _Float16 As[2][BM * BK];   // [m_local][k_local]
  __shared__ _Float16 Bt[2][BN * BK];   // [n_local][k_local] (transposed im2col)

  const int t    = threadIdx.x;
  const int lane = t & 31;
  const int wave = t >> 5;           // 0..7
  const int wm   = wave >> 1;        // 0..3 -> M offset wm*32
  const int wn   = wave & 1;         // 0..1 -> N offset wn*32
  const int mBlock = blockIdx.y * BM;
  const int nBlock = blockIdx.x * BN;

  // A staging map (non-TDM fallback): 128 rows x 32 k, 16 halves per thread
  const int arow = t >> 1;           // 0..127
  const int acol = (t & 1) * 16;     // 0,16
  const _Float16* agp = wf + (mBlock + arow) * KSZ + acol;

  // B staging map: 64 n x 32 k, 8 halves per thread, k wave-uniform
  const int bn   = t & 63;           // consecutive threads -> consecutive n
  const int bk0  = (t >> 6) * 8;     // 0,8,16,24 (uniform within a wave)
  const int bk0u = __builtin_amdgcn_readfirstlane(bk0);
  const int ng  = nBlock + bn;
  const int bb  = ng / HW;
  const int sp  = ng - bb * HW;      // oy*56+ox
  const int oy  = sp / HSZ;
  const int ox  = sp - oy * HSZ;
  const _Float16* xbase = xf + (size_t)(bb * C_IN) * HW;
  const int zoff = XTOT - bb * C_IN * HW;   // index of zero pad relative to xbase

  // fragment index maps (contiguous k order K=16*(lane/16)+s, same for A and B)
  const int ml  = lane & 15;
  const int kbA = (lane >> 4) * 16;  // A: halves h -> K=kbA+h
  const int kbB = (lane >> 4) * 16;  // B: halves h -> K=kbB+h

  v8f acc[2][2] = {};

  auto stage = [&](int buf, int kstep) {
    // ---- A tile: 2D tile 128 x 32 halves, stride 2304 ----
#ifdef HAVE_TDM
    if (wave == 0) {
      unsigned long long ga =
          (unsigned long long)(uintptr_t)(wf + (size_t)mBlock * KSZ + kstep);
      unsigned int ldsa =
          (unsigned int)(uintptr_t)(AS3 _Float16*)&As[buf][0];
      v4u g0 = { 1u,                                   // count=1, user mode
                 ldsa,                                 // lds_addr (bytes)
                 (unsigned int)(ga & 0xffffffffu),     // global_addr[31:0]
                 (unsigned int)((ga >> 32) & 0x1ffffffu) | (2u << 30) }; // [56:32]+type
      v8i g1 = { 0x10000,                              // data_size=1 (2B), mask=0
                 0x09000000,                           // tensor_dim0=2304 -> bits[79:48]
                 0,                                    // tensor_dim0 hi16 / dim1 lo16
                 0x10 | (32 << 16),                    // dim1 hi16 (dim1=1<<20), tile_dim0=32
                 128,                                  // tile_dim1=128
                 2304,                                 // tensor_dim0_stride lo32
                 0, 0 };
      v4i z4 = { 0, 0, 0, 0 };
#if __clang_major__ >= 23
      v8i z8 = { 0, 0, 0, 0, 0, 0, 0, 0 };
      __builtin_amdgcn_tensor_load_to_lds(g0, g1, z4, z4, z8, 0);
#else
      __builtin_amdgcn_tensor_load_to_lds(g0, g1, z4, z4, 0);
#endif
    }
#elif defined(HAVE_ASYNC_LDS)
    const AS1 _Float16* gAh = (const AS1 _Float16*)(agp + kstep);
    AS3 _Float16* lAh = (AS3 _Float16*)&As[buf][arow * BK + acol];
    AS1 v4i* gA = (AS1 v4i*)gAh;
    AS3 v4i* lA = (AS3 v4i*)lAh;
    __builtin_amdgcn_global_load_async_to_lds_b128(gA, lA, 0, 0);
    __builtin_amdgcn_global_load_async_to_lds_b128(gA + 1, lA + 1, 0, 0);
#else
    *(v8h*)(&As[buf][arow * BK + acol])     = *(const v8h*)(agp + kstep);
    *(v8h*)(&As[buf][arow * BK + acol + 8]) = *(const v8h*)(agp + kstep + 8);
#endif
    // ---- B tile: im2col gather; k decomposition is wave-uniform (SALU) ----
    v8h bv;
#pragma unroll
    for (int j = 0; j < 8; ++j) {
      int k    = kstep + bk0u + j;        // uniform
      int c    = k / 9;                   // SALU div
      int rs   = k - c * 9;
      int kh   = rs / 3;
      int kw   = rs - kh * 3;
      int uoff = c * HW + (kh - 1) * HSZ + (kw - 1);   // uniform offset
      int yi   = oy + (kh - 1);
      int xi   = ox + (kw - 1);
      bool ok  = ((unsigned)yi < (unsigned)HSZ) & ((unsigned)xi < (unsigned)HSZ);
      int idx  = ok ? (uoff + sp) : zoff;              // zoff -> zero pad
      bv[j] = xbase[idx];
    }
    *(v8h*)(&Bt[buf][bn * BK + bk0]) = bv;
  };

  auto wait_stage = [&]() {
#ifdef HAVE_TDM
    __builtin_amdgcn_s_wait_tensorcnt(0);
#elif defined(HAVE_ASYNC_LDS)
    asm volatile("s_wait_asynccnt 0x0" ::: "memory");
#endif
  };

  stage(0, 0);
  wait_stage();
  __syncthreads();

  for (int s = 0; s < NSTEPS; ++s) {
    const int cur = s & 1;
    if (s + 1 < NSTEPS) stage(cur ^ 1, (s + 1) * BK);

    const _Float16* A_ = As[cur];
    const _Float16* B_ = Bt[cur];
    v16h afrag[2], bfrag[2];
#pragma unroll
    for (int im = 0; im < 2; ++im) {
      const _Float16* ap = A_ + (wm * 32 + im * 16 + ml) * BK;
      v8h lo = *(const v8h*)(ap + kbA);
      v8h hi = *(const v8h*)(ap + kbA + 8);
      afrag[im] = __builtin_shufflevector(lo, hi,
          0,1,2,3,4,5,6,7,8,9,10,11,12,13,14,15);
    }
#pragma unroll
    for (int in_ = 0; in_ < 2; ++in_) {
      const _Float16* bp = B_ + (wn * 32 + in_ * 16 + ml) * BK + kbB;
      v8h lo = *(const v8h*)(bp);
      v8h hi = *(const v8h*)(bp + 8);
      bfrag[in_] = __builtin_shufflevector(lo, hi,
          0,1,2,3,4,5,6,7,8,9,10,11,12,13,14,15);
    }
#pragma unroll
    for (int im = 0; im < 2; ++im)
#pragma unroll
      for (int in_ = 0; in_ < 2; ++in_)
        acc[im][in_] = __builtin_amdgcn_wmma_f32_16x16x32_f16(
            false, afrag[im], false, bfrag[in_],
            (short)0, acc[im][in_], false, false);

    wait_stage();
    __syncthreads();
  }

  // ---- store C: per acc, lane layout VGPR r -> M = r + (lane>=16)*8, N = lane&15
  const int rowoff = (lane >> 4) * 8;
#pragma unroll
  for (int in_ = 0; in_ < 2; ++in_) {
    const int n   = nBlock + wn * 32 + in_ * 16 + ml;
    const int b_  = n / HW;
    const int sp_ = n - b_ * HW;
    float* obase = out + (size_t)b_ * K_OUT * HW + sp_;
#pragma unroll
    for (int im = 0; im < 2; ++im) {
      const int o0 = mBlock + wm * 32 + im * 16 + rowoff;
#pragma unroll
      for (int r = 0; r < 8; ++r)
        obase[(size_t)(o0 + r) * HW] = acc[im][in_][r];
    }
  }
}

// ---------------- launcher ----------------
extern "C" void kernel_launch(void* const* d_in, const int* in_sizes, int n_in,
                              void* d_out, int out_size, void* d_ws, size_t ws_size,
                              hipStream_t stream) {
  (void)in_sizes; (void)n_in; (void)out_size; (void)ws_size;
  const float* x   = (const float*)d_in[0];
  const float* w   = (const float*)d_in[1];
  const float* cb  = (const float*)d_in[2];
  const int*   sig = (const int*)d_in[3];
  float* out = (float*)d_out;

  char* ws = (char*)d_ws;
  float*    stats = (float*)ws;                        // 8 B
  _Float16* wf    = (_Float16*)(ws + 64);              // 589824 halves
  _Float16* xf    = (_Float16*)(ws + 64 + 1179648);    // 25690112 halves + 16 pad

  init_ws<<<1, 1, 0, stream>>>(stats, xf + XTOT);
  w_stats<<<NGRP / 256, 256, 0, stream>>>(w, stats);
  w_final<<<NGRP / 256, 256, 0, stream>>>(w, cb, sig, stats, wf);
  x_to_f16<<<(XTOT / 4 + 255) / 256, 256, 0, stream>>>(x, xf, XTOT / 4);

  dim3 grid(NTOT / BN, K_OUT / BM);
  conv_wmma<<<grid, 256, 0, stream>>>(xf, wf, out);
}
